// SpatioTemporalFusionBlock_17497696764293
// MI455X (gfx1250) — matrix operand order repaired and checked by reading, not compiled, not run on hardware
//
#include <hip/hip_runtime.h>
#include <hip/hip_bf16.h>

typedef __attribute__((ext_vector_type(16))) _Float16 v16h;
typedef __attribute__((ext_vector_type(8)))  float    v8f;

#define N_NODES 50000
#define T_STEPS 24
#define D_IN    32
#define D_HID   64
#define EPS_LN  1e-5f

static __device__ inline v8f vz8() {
  v8f z = {0.f,0.f,0.f,0.f,0.f,0.f,0.f,0.f};
  return z;
}

// -------------------------------------------------------------------------
// Kernel 1: HW_f16[N*T][64] = f16( H[N*T][32] @ W[32][64] )
// One wave (32 lanes) per 16-row tile.  A per ISA 16x32 f16 A-layout,
// B = W staged in LDS as per-lane-contiguous B fragments (4 N-tiles).
// -------------------------------------------------------------------------
__global__ __launch_bounds__(256) void k_gemm_in(
    const float* __restrict__ H, const float* __restrict__ W,
    _Float16* __restrict__ HW, int ntiles)
{
  __shared__ _Float16 sW[4 * 32 * 16];   // 4 fragments x 32 lanes x 16 halfs
  for (int e = threadIdx.x; e < 2048; e += 256) {
    int j    = e >> 9;
    int lane = (e >> 4) & 31;
    int q    = e & 15;
    int K    = ((lane & 16) ? 8 : 0) + (q & 7) + ((q & 8) ? 16 : 0);  // B: K in table order (8h+s, 16+8h+s-8)
    int co   = j * 16 + (lane & 15);                // N striped across lanes
    sW[e] = (_Float16)W[K * 64 + co];
  }
  __syncthreads();

  int wave = blockIdx.x * 8 + (threadIdx.x >> 5);
  if (wave >= ntiles) return;
  int lane = threadIdx.x & 31;
  int m  = lane & 15;
  int hi = lane >> 4;
  int kb = hi * 8;
  int nloc = lane & 15;

  size_t row = (size_t)wave * 16 + m;
  const float* hp = H + row * D_IN + kb;
  float4 x0 = *(const float4*)(hp);
  float4 x1 = *(const float4*)(hp + 4);
  float4 x2 = *(const float4*)(hp + 16);
  float4 x3 = *(const float4*)(hp + 20);

  v16h a;
  a[0]=(_Float16)x0.x; a[1]=(_Float16)x0.y; a[2]=(_Float16)x0.z; a[3]=(_Float16)x0.w;
  a[4]=(_Float16)x1.x; a[5]=(_Float16)x1.y; a[6]=(_Float16)x1.z; a[7]=(_Float16)x1.w;
  a[8]=(_Float16)x2.x; a[9]=(_Float16)x2.y; a[10]=(_Float16)x2.z; a[11]=(_Float16)x2.w;
  a[12]=(_Float16)x3.x; a[13]=(_Float16)x3.y; a[14]=(_Float16)x3.z; a[15]=(_Float16)x3.w;

#pragma unroll
  for (int j = 0; j < 4; ++j) {
    v16h bf = *(const v16h*)&sW[(j * 32 + lane) * 16];
    v8f c = vz8();
    c = __builtin_amdgcn_wmma_f32_16x16x32_f16(false, a, false, bf,
                                               (short)0, c, false, false);
#pragma unroll
    for (int r = 0; r < 8; ++r) {
      size_t orow = (size_t)wave * 16 + r + 8 * hi;     // C/D: M = r + 8*(lane>=16)
      HW[orow * 64 + j * 16 + nloc] = (_Float16)c[r];
    }
  }
}

// -------------------------------------------------------------------------
// Kernel 2: SpMM scatter.  One wave per edge: gather f16 row (T*64 = 1536
// halfs) of src with coalesced b64 loads, fp32 atomic-add into GC[dst].
// -------------------------------------------------------------------------
__global__ __launch_bounds__(256) void k_spmm(
    const _Float16* __restrict__ HW, const int* __restrict__ esrc,
    const int* __restrict__ edst, const float* __restrict__ ev,
    float* __restrict__ GC, int E)
{
  int e = blockIdx.x * 8 + (threadIdx.x >> 5);
  if (e >= E) return;
  int lane = threadIdx.x & 31;
  int src  = esrc[e];
  int dst  = edst[e];
  float val = ev[e];

  const uint2* sp = (const uint2*)(HW + (size_t)src * (T_STEPS * D_HID));
  float* dp = GC + (size_t)dst * (T_STEPS * D_HID);

#pragma unroll
  for (int i = 0; i < 12; ++i) {            // 12 * 32 lanes * 4 halfs = 1536
    int idx = i * 32 + lane;
    union { uint2 u; _Float16 h[4]; } cv;
    cv.u = sp[idx];
    float* q = dp + idx * 4;
    atomicAdd(q + 0, (float)cv.h[0] * val);
    atomicAdd(q + 1, (float)cv.h[1] * val);
    atomicAdd(q + 2, (float)cv.h[2] * val);
    atomicAdd(q + 3, (float)cv.h[3] * val);
  }
}

// -------------------------------------------------------------------------
// Kernel 3 (fused): bias+ReLU -> causal gated conv (3 t-shifted 64x64 WMMA
// GEMMs, f & g) -> tanh*sigmoid -> residual (H @ res_w via WMMA) -> LayerNorm.
// One wave per (16-node tile, t).  48 conv WMMAs + 4 residual WMMAs / wave.
// -------------------------------------------------------------------------
__global__ __launch_bounds__(256) void k_fuse(
    const float* __restrict__ GC, const float* __restrict__ H,
    const float* __restrict__ conv_f_w, const float* __restrict__ conv_f_b,
    const float* __restrict__ conv_g_w, const float* __restrict__ conv_g_b,
    const float* __restrict__ res_w, const float* __restrict__ res_b,
    const float* __restrict__ bias, const float* __restrict__ ln_g,
    const float* __restrict__ ln_b, float* __restrict__ out, int nwaves)
{
  // frag id = ((conv*3 + k)*4 + j)*2 + chunk  (48 fragments)
  __shared__ _Float16 sWgt[48 * 32 * 16];   // 49152 B
  __shared__ _Float16 sRes[4 * 32 * 16];    //  4096 B
  __shared__ float sB[64], sCF[64], sCG[64], sRB[64], sLG[64], sLB[64];

  for (int e = threadIdx.x; e < 48 * 512; e += 256) {
    int frag = e >> 9;
    int lane = (e >> 4) & 31;
    int q    = e & 15;
    int chunk = frag & 1;
    int j     = (frag >> 1) & 3;
    int kc    = frag >> 3;
    int k     = kc % 3;
    int conv  = kc / 3;
    int ci = chunk * 32 + ((lane & 16) ? 8 : 0) + (q & 7) + ((q & 8) ? 16 : 0);   // K dim = input channel
    int co = j * 16 + (lane & 15);                      // N dim = output channel
    const float* wp = conv ? conv_g_w : conv_f_w;       // [out][in][K]
    sWgt[e] = (_Float16)wp[(co * 64 + ci) * 3 + k];
  }
  for (int e = threadIdx.x; e < 2048; e += 256) {
    int j    = e >> 9;
    int lane = (e >> 4) & 31;
    int q    = e & 15;
    int K    = ((lane & 16) ? 8 : 0) + (q & 7) + ((q & 8) ? 16 : 0);
    int co   = j * 16 + (lane & 15);
    sRes[e] = (_Float16)res_w[K * 64 + co];             // res_w [32][64]
  }
  if (threadIdx.x < 64) {
    int i = threadIdx.x;
    sB[i] = bias[i];   sCF[i] = conv_f_b[i]; sCG[i] = conv_g_b[i];
    sRB[i] = res_b[i]; sLG[i] = ln_g[i];     sLB[i] = ln_b[i];
  }
  __syncthreads();

  int wave = blockIdx.x * 8 + (threadIdx.x >> 5);
  if (wave >= nwaves) return;
  int lane = threadIdx.x & 31;
  int m    = lane & 15;
  int hi   = lane >> 4;
  int kb   = hi * 8;
  int nloc = lane & 15;

  int t     = wave % T_STEPS;
  int node0 = (wave / T_STEPS) * 16;
  size_t nodeRow = (size_t)(node0 + m);

  v8f accF[4], accG[4], accR[4];
#pragma unroll
  for (int j = 0; j < 4; ++j) { accF[j] = vz8(); accG[j] = vz8(); accR[j] = vz8(); }

  // ---- residual projection: A = f16(H[node][t][0:32]) ----
  {
    const float* hp = H + (nodeRow * T_STEPS + t) * D_IN + kb;
    float4 x0 = *(const float4*)(hp);
    float4 x1 = *(const float4*)(hp + 4);
    float4 x2 = *(const float4*)(hp + 16);
    float4 x3 = *(const float4*)(hp + 20);
    v16h aR;
    aR[0]=(_Float16)x0.x; aR[1]=(_Float16)x0.y; aR[2]=(_Float16)x0.z; aR[3]=(_Float16)x0.w;
    aR[4]=(_Float16)x1.x; aR[5]=(_Float16)x1.y; aR[6]=(_Float16)x1.z; aR[7]=(_Float16)x1.w;
    aR[8]=(_Float16)x2.x; aR[9]=(_Float16)x2.y; aR[10]=(_Float16)x2.z; aR[11]=(_Float16)x2.w;
    aR[12]=(_Float16)x3.x; aR[13]=(_Float16)x3.y; aR[14]=(_Float16)x3.z; aR[15]=(_Float16)x3.w;
#pragma unroll
    for (int j = 0; j < 4; ++j) {
      v16h bR = *(const v16h*)&sRes[(j * 32 + lane) * 16];
      accR[j] = __builtin_amdgcn_wmma_f32_16x16x32_f16(false, aR, false, bR,
                                                       (short)0, accR[j], false, false);
    }
  }

  // ---- causal gated conv: 3 t-shifted GEMMs over K=64 (2 chunks of 32) ----
#pragma unroll
  for (int k = 0; k < 3; ++k) {
    int tsrc = t - 2 + k;
#pragma unroll
    for (int chunk = 0; chunk < 2; ++chunk) {
      v16h a;
      if (tsrc >= 0) {
        const float* gp = GC + (nodeRow * T_STEPS + tsrc) * D_HID + chunk * 32 + kb;
        int cb = chunk * 32 + kb;
        float4 y0 = *(const float4*)(gp);
        float4 y1 = *(const float4*)(gp + 4);
        float4 y2 = *(const float4*)(gp + 16);
        float4 y3 = *(const float4*)(gp + 20);
        a[0] = (_Float16)fmaxf(y0.x + sB[cb + 0], 0.f);
        a[1] = (_Float16)fmaxf(y0.y + sB[cb + 1], 0.f);
        a[2] = (_Float16)fmaxf(y0.z + sB[cb + 2], 0.f);
        a[3] = (_Float16)fmaxf(y0.w + sB[cb + 3], 0.f);
        a[4] = (_Float16)fmaxf(y1.x + sB[cb + 4], 0.f);
        a[5] = (_Float16)fmaxf(y1.y + sB[cb + 5], 0.f);
        a[6] = (_Float16)fmaxf(y1.z + sB[cb + 6], 0.f);
        a[7] = (_Float16)fmaxf(y1.w + sB[cb + 7], 0.f);
        a[8]  = (_Float16)fmaxf(y2.x + sB[cb + 16], 0.f);
        a[9]  = (_Float16)fmaxf(y2.y + sB[cb + 17], 0.f);
        a[10] = (_Float16)fmaxf(y2.z + sB[cb + 18], 0.f);
        a[11] = (_Float16)fmaxf(y2.w + sB[cb + 19], 0.f);
        a[12] = (_Float16)fmaxf(y3.x + sB[cb + 20], 0.f);
        a[13] = (_Float16)fmaxf(y3.y + sB[cb + 21], 0.f);
        a[14] = (_Float16)fmaxf(y3.z + sB[cb + 22], 0.f);
        a[15] = (_Float16)fmaxf(y3.w + sB[cb + 23], 0.f);
      } else {
#pragma unroll
        for (int q = 0; q < 16; ++q) a[q] = (_Float16)0.f;
      }
#pragma unroll
      for (int j = 0; j < 4; ++j) {
        int ff = (((0 * 3 + k) * 4 + j) * 2 + chunk);
        v16h bf = *(const v16h*)&sWgt[(ff * 32 + lane) * 16];
        accF[j] = __builtin_amdgcn_wmma_f32_16x16x32_f16(false, a, false, bf,
                                                         (short)0, accF[j], false, false);
        int fg = (((1 * 3 + k) * 4 + j) * 2 + chunk);
        v16h bg = *(const v16h*)&sWgt[(fg * 32 + lane) * 16];
        accG[j] = __builtin_amdgcn_wmma_f32_16x16x32_f16(false, a, false, bg,
                                                         (short)0, accG[j], false, false);
      }
    }
  }

  // ---- gate + residual, then LayerNorm per 64-wide row ----
  float yv[4][8];
#pragma unroll
  for (int j = 0; j < 4; ++j) {
    int col = j * 16 + nloc;
#pragma unroll
    for (int r = 0; r < 8; ++r) {
      float fv = tanhf(accF[j][r] + sCF[col]);
      float gz = accG[j][r] + sCG[col];
      float gv = 1.f / (1.f + expf(-gz));
      yv[j][r] = fv * gv + accR[j][r] + sRB[col];
    }
  }

#pragma unroll
  for (int r = 0; r < 8; ++r) {
    float s  = yv[0][r] + yv[1][r] + yv[2][r] + yv[3][r];
    float ss = yv[0][r]*yv[0][r] + yv[1][r]*yv[1][r]
             + yv[2][r]*yv[2][r] + yv[3][r]*yv[3][r];
#pragma unroll
    for (int msk = 1; msk < 16; msk <<= 1) {   // rows live in 16-lane halves
      s  += __shfl_xor(s,  msk, 32);
      ss += __shfl_xor(ss, msk, 32);
    }
    float mu  = s * (1.f / 64.f);
    float var = ss * (1.f / 64.f) - mu * mu;
    float inv = rsqrtf(var + EPS_LN);
    size_t orow = ((size_t)(node0 + r + 8 * hi) * T_STEPS + t) * D_HID;
#pragma unroll
    for (int j = 0; j < 4; ++j) {
      int col = j * 16 + nloc;
      out[orow + col] = (yv[j][r] - mu) * inv * sLG[col] + sLB[col];
    }
  }
}

// -------------------------------------------------------------------------
extern "C" void kernel_launch(void* const* d_in, const int* in_sizes, int n_in,
                              void* d_out, int out_size, void* d_ws, size_t ws_size,
                              hipStream_t stream)
{
  const float* H    = (const float*)d_in[0];
  const int*   esrc = (const int*)  d_in[1];
  const int*   edst = (const int*)  d_in[2];
  const float* ev   = (const float*)d_in[3];
  const float* W    = (const float*)d_in[4];
  const float* b    = (const float*)d_in[5];
  const float* cfw  = (const float*)d_in[6];
  const float* cfb  = (const float*)d_in[7];
  const float* cgw  = (const float*)d_in[8];
  const float* cgb  = (const float*)d_in[9];
  const float* rw   = (const float*)d_in[10];
  const float* rb   = (const float*)d_in[11];
  const float* lg   = (const float*)d_in[12];
  const float* lb   = (const float*)d_in[13];
  int E = in_sizes[1];

  const size_t NT = (size_t)N_NODES * T_STEPS;
  _Float16* HW = (_Float16*)d_ws;                               // 153.6 MB f16
  float*    GC = (float*)((char*)d_ws + NT * 64 * sizeof(_Float16)); // 307.2 MB f32

  int ntiles = (int)(NT / 16);                  // 75000, exact
  k_gemm_in<<<(ntiles + 7) / 8, 256, 0, stream>>>(H, W, HW, ntiles);

  hipMemsetAsync(GC, 0, NT * 64 * sizeof(float), stream);

  k_spmm<<<(E + 7) / 8, 256, 0, stream>>>(HW, esrc, edst, ev, GC, E);

  int nwaves = (N_NODES / 16) * T_STEPS;        // 75000, exact
  k_fuse<<<(nwaves + 7) / 8, 256, 0, stream>>>(GC, H, cfw, cfb, cgw, cgb,
                                               rw, rb, b, lg, lb,
                                               (float*)d_out, nwaves);
}
